// MoEProcessor_11003706213302
// MI455X (gfx1250) — matrix operand order repaired and checked by reading, not compiled, not run on hardware
//
#include <hip/hip_runtime.h>
#include <hip/hip_bf16.h>

typedef _Float16 h16;
typedef __attribute__((ext_vector_type(16))) _Float16 v16h;
typedef __attribute__((ext_vector_type(8)))  _Float16 v8h;
typedef __attribute__((ext_vector_type(8)))  float    v8f;
typedef __attribute__((ext_vector_type(4)))  int      v4i;

#define EMBED    256
#define TEXT_D   768
#define N_TOK    1024
#define BATCH    16
#define M_TOK    (BATCH*N_TOK)     /* 16384 */
#define HEADS    4
#define HEAD_DIM 64
#define MLP_HID  1024
#define KNO_LIFT 512
#define N_MODE_R 256               /* 16 kx * 8 ky * 2 (re,im) */

#define BM 128
#define BN 128
#define KC 32
#define LDS_PAD 8

// CDNA5 async global->LDS copy path (ASYNCcnt), with sync fallback.
#if defined(__AMDGCN__) && __has_builtin(__builtin_amdgcn_global_load_async_to_lds_b128)
#define ASYNC_LDS 1
typedef v4i __attribute__((address_space(1))) as1_v4i;
typedef v4i __attribute__((address_space(3))) as3_v4i;
#else
#define ASYNC_LDS 0
#endif

__device__ __forceinline__ void async_g2l_b128(const void* g, void* l) {
#if ASYNC_LDS
    __builtin_amdgcn_global_load_async_to_lds_b128(
        (as1_v4i*)g, (as3_v4i*)l, 0, 0);
#else
    (void)g; (void)l;
#endif
}

__device__ __forceinline__ void wait_async_pending(int pending) {
#if ASYNC_LDS
#if __has_builtin(__builtin_amdgcn_s_wait_asynccnt)
    if (pending) __builtin_amdgcn_s_wait_asynccnt(4);
    else         __builtin_amdgcn_s_wait_asynccnt(0);
#else
    if (pending) asm volatile("s_wait_asynccnt 0x4" ::: "memory");
    else         asm volatile("s_wait_asynccnt 0x0" ::: "memory");
#endif
#endif
}

// ---------------------------------------------------------------- utilities
__device__ __forceinline__ float gelu_f(float x) {
    return 0.5f * x * (1.0f + erff(x * 0.7071067811865475f));
}

__device__ __forceinline__ v16h frag_cat(const h16* p, int o0, int o1) {
    v8h lo = *(const v8h*)(p + o0);
    v8h hi = *(const v8h*)(p + o1);
    return __builtin_shufflevector(lo, hi, 0,1,2,3,4,5,6,7,8,9,10,11,12,13,14,15);
}

// ---------------------------------------------------------------- WMMA GEMM
// C[m][n] = act( alpha * sum_k A[m][k]*Bt[n][k] + bias[n] + resid ) ; batched
// via blockIdx.z -> (zb,zh) with per-axis strides. A,Bt are f16; Cf f32 and/or
// Ch f16 outputs (either may be null). act: 0=none 1=gelu.
// Full tiles stream through the CDNA5 async global->LDS pipe, double-buffered;
// edge tiles fall back to guarded VGPR staging.
__global__ __launch_bounds__(256) void wmma_gemm_kernel(
    const h16* __restrict__ A, const h16* __restrict__ Bt,
    const float* __restrict__ bias, const float* __restrict__ resid,
    float* __restrict__ Cf, h16* __restrict__ Ch,
    int M, int N, int K, int lda, int ldb, int ldc,
    long sAb, long sAh, long sBb, long sBh, long sCb, long sCh,
    int hdiv, float alpha, int act)
{
    __shared__ h16 As[2][BM][KC + LDS_PAD];
    __shared__ h16 Bs[2][BN][KC + LDS_PAD];

    int z  = blockIdx.z;
    int zb = z / hdiv, zh = z - zb * hdiv;
    A  += (long)zb * sAb + (long)zh * sAh;
    Bt += (long)zb * sBb + (long)zh * sBh;
    long coff = (long)zb * sCb + (long)zh * sCh;

    int m0 = blockIdx.x * BM, n0 = blockIdx.y * BN;
    int t = threadIdx.x, lane = t & 31, wave = t >> 5;
    int wm = wave >> 1, wn = wave & 1;            // waves: 4 (rows) x 2 (cols)
    int lm = lane & 15, lhalf = lane >> 4;

    v8f zero = {0.f,0.f,0.f,0.f,0.f,0.f,0.f,0.f};
    v8f acc[2][4];
    #pragma unroll
    for (int i = 0; i < 2; ++i)
        #pragma unroll
        for (int j = 0; j < 4; ++j) acc[i][j] = zero;

    // chunk coordinates shared by loaders: 512 16B-chunks per 128x32 tile
    int cr0 = t >> 2,            cc0 = (t & 3) * 8;
    int cr1 = (t + 256) >> 2,    cc1 = ((t + 256) & 3) * 8;

    auto compute_tile = [&](int buf) {
        int akb = lhalf * 8;      // A frag: lanes 0-15 K{0..7,16..23}; 16-31 K{8..15,24..31}
        v16h a0 = frag_cat(&As[buf][wm * 32 +  0 + lm][0], akb, akb + 16);
        v16h a1 = frag_cat(&As[buf][wm * 32 + 16 + lm][0], akb, akb + 16);
        int bkb = lhalf * 8;      // B frag: lanes 0-15 K{0..7,16..23}; 16-31 K{8..15,24..31}
        #pragma unroll
        for (int j = 0; j < 4; ++j) {
            v16h b = frag_cat(&Bs[buf][wn * 64 + j * 16 + lm][0], bkb, bkb + 16);
            acc[0][j] = __builtin_amdgcn_wmma_f32_16x16x32_f16(false, a0, false, b, (short)0, acc[0][j], false, false);
            acc[1][j] = __builtin_amdgcn_wmma_f32_16x16x32_f16(false, a1, false, b, (short)0, acc[1][j], false, false);
        }
    };

    bool fullTile = (m0 + BM <= M) && (n0 + BN <= N);

    if (ASYNC_LDS && fullTile) {
        // 4 async instructions per tile (2 for A, 2 for B), double-buffered.
        auto issue_tile = [&](int k0, int buf) {
            async_g2l_b128(A  + (long)(m0 + cr0) * lda + (k0 + cc0), &As[buf][cr0][cc0]);
            async_g2l_b128(A  + (long)(m0 + cr1) * lda + (k0 + cc1), &As[buf][cr1][cc1]);
            async_g2l_b128(Bt + (long)(n0 + cr0) * ldb + (k0 + cc0), &Bs[buf][cr0][cc0]);
            async_g2l_b128(Bt + (long)(n0 + cr1) * ldb + (k0 + cc1), &Bs[buf][cr1][cc1]);
        };
        int ntiles = K / KC;
        issue_tile(0, 0);
        for (int it = 0; it < ntiles; ++it) {
            int cur = it & 1;
            if (it + 1 < ntiles) {
                issue_tile((it + 1) * KC, cur ^ 1);
                wait_async_pending(1);            // own current tile has landed
            } else {
                wait_async_pending(0);
            }
            __syncthreads();                      // all waves' copies visible
            compute_tile(cur);
            __syncthreads();                      // done reading before overwrite
        }
    } else {
        for (int k0 = 0; k0 < K; k0 += KC) {
            {
                int gm = m0 + cr0;
                v8h va = {};
                if (gm < M) va = *(const v8h*)(A + (long)gm * lda + (k0 + cc0));
                *(v8h*)(&As[0][cr0][cc0]) = va;
            }
            {
                int gm = m0 + cr1;
                v8h va = {};
                if (gm < M) va = *(const v8h*)(A + (long)gm * lda + (k0 + cc1));
                *(v8h*)(&As[0][cr1][cc1]) = va;
            }
            {
                int gn = n0 + cr0;
                v8h vb = {};
                if (gn < N) vb = *(const v8h*)(Bt + (long)gn * ldb + (k0 + cc0));
                *(v8h*)(&Bs[0][cr0][cc0]) = vb;
            }
            {
                int gn = n0 + cr1;
                v8h vb = {};
                if (gn < N) vb = *(const v8h*)(Bt + (long)gn * ldb + (k0 + cc1));
                *(v8h*)(&Bs[0][cr1][cc1]) = vb;
            }
            if (k0 + KC < K) {                    // global_prefetch_b8 next tile
                int pr = m0 + (t >> 1);
                if (pr < M) __builtin_prefetch(A + (long)pr * lda + (k0 + KC), 0, 0);
            }
            __syncthreads();
            compute_tile(0);
            __syncthreads();
        }
    }

    #pragma unroll
    for (int im = 0; im < 2; ++im) {
        #pragma unroll
        for (int jn = 0; jn < 4; ++jn) {
            int mt = m0 + wm * 32 + im * 16;
            int nt = n0 + wn * 64 + jn * 16;
            int n  = nt + lm;
            float bi = (bias && n < N) ? bias[n] : 0.0f;
            #pragma unroll
            for (int r = 0; r < 8; ++r) {
                int m = mt + r + (lhalf << 3);    // VGPR r: lanes0-15 M=r, 16-31 M=r+8
                if (m < M && n < N) {
                    long o = coff + (long)m * ldc + n;
                    float v = acc[im][jn][r] * alpha + bi;
                    if (resid) v += resid[o];
                    if (act == 1) v = gelu_f(v);
                    if (Cf) Cf[o] = v;
                    if (Ch) Ch[o] = (h16)v;
                }
            }
        }
    }
}

// ---------------------------------------------------------------- small kernels
__global__ void copy_f32_kernel(float* dst, const float* src, int n) {
    int i = blockIdx.x * blockDim.x + threadIdx.x;
    if (i < n) dst[i] = src[i];
}

__global__ void wtconv_kernel(const float* __restrict__ W, h16* __restrict__ Wt,
                              int K, int N) {
    int i = blockIdx.x * blockDim.x + threadIdx.x;
    if (i >= K * N) return;
    int k = i / N, n = i - k * N;
    Wt[(long)n * K + k] = (h16)W[i];
}

// x mean over tokens + text -> relu fc -> logits -> top2 softmax dense weights
__global__ void gate_kernel(const float* __restrict__ x, const float* __restrict__ txt,
                            const float* __restrict__ w1, const float* __restrict__ b1,
                            const float* __restrict__ w2, const float* __restrict__ b2,
                            float* __restrict__ gatew) {
    int b = blockIdx.x, t = threadIdx.x;
    __shared__ float feat[EMBED + TEXT_D];
    __shared__ float hh[EMBED];
    __shared__ float lg[4];
    const float* xb = x + (long)b * N_TOK * EMBED;
    float s = 0.f;
    for (int n = 0; n < N_TOK; ++n) s += xb[(long)n * EMBED + t];
    feat[t] = s * (1.0f / N_TOK);
    for (int i = t; i < TEXT_D; i += EMBED) feat[EMBED + i] = txt[(long)b * TEXT_D + i];
    __syncthreads();
    float a = b1[t];
    for (int k = 0; k < EMBED + TEXT_D; ++k) a += feat[k] * w1[(long)k * EMBED + t];
    hh[t] = fmaxf(a, 0.0f);
    __syncthreads();
    if (t < 4) {
        float l = b2[t];
        for (int j = 0; j < EMBED; ++j) l += hh[j] * w2[j * 4 + t];
        lg[t] = l;
    }
    __syncthreads();
    if (t == 0) {
        int i0 = 0; float v0 = lg[0];
        for (int i = 1; i < 4; ++i) if (lg[i] > v0) { v0 = lg[i]; i0 = i; }
        int i1 = -1; float v1 = -3.0e38f;
        for (int i = 0; i < 4; ++i) { if (i == i0) continue; if (lg[i] > v1) { v1 = lg[i]; i1 = i; } }
        float e0 = 1.0f, e1 = expf(v1 - v0);
        float den = e0 + e1;
        float w[4] = {0.f,0.f,0.f,0.f};
        w[i0] = e0 / den; w[i1] = e1 / den;
        for (int i = 0; i < 4; ++i) gatew[b * 4 + i] = w[i];
    }
}

// LayerNorm over last dim 256; writes f16 for the next GEMM
__global__ void ln_kernel(const float* __restrict__ x, const float* __restrict__ g,
                          const float* __restrict__ bb, h16* __restrict__ out) {
    int row = blockIdx.x, t = threadIdx.x;
    __shared__ float red[EMBED];
    __shared__ float smean, svar;
    float v = x[(long)row * EMBED + t];
    red[t] = v; __syncthreads();
    for (int s = EMBED / 2; s > 0; s >>= 1) { if (t < s) red[t] += red[t + s]; __syncthreads(); }
    if (t == 0) smean = red[0] * (1.0f / EMBED);
    __syncthreads();
    float d = v - smean;
    red[t] = d * d; __syncthreads();
    for (int s = EMBED / 2; s > 0; s >>= 1) { if (t < s) red[t] += red[t + s]; __syncthreads(); }
    if (t == 0) svar = red[0] * (1.0f / EMBED);
    __syncthreads();
    float r = rsqrtf(svar + 1e-5f);
    out[(long)row * EMBED + t] = (h16)(d * r * g[t] + bb[t]);
}

// per-head LN of k,v (64-wide) from f16 qkv; writes kT/vT transposed: [b][h][d][n]
__global__ void headln_kernel(const h16* __restrict__ qkv,
                              const float* __restrict__ kg, const float* __restrict__ kb,
                              const float* __restrict__ vg, const float* __restrict__ vb,
                              h16* __restrict__ kT, h16* __restrict__ vT) {
    int idx = blockIdx.x * blockDim.x + threadIdx.x;
    if (idx >= BATCH * N_TOK * HEADS) return;
    int h = idx & 3, n = (idx >> 2) & (N_TOK - 1), b = idx >> 12;
    const h16* row = qkv + ((long)(b * N_TOK + n)) * (3 * EMBED);
    for (int which = 0; which < 2; ++which) {
        const h16* src = row + (which ? 2 * EMBED : EMBED) + h * HEAD_DIM;
        const float* g  = which ? vg : kg;
        const float* be = which ? vb : kb;
        h16* dst = which ? vT : kT;
        float m = 0.f;
        for (int d = 0; d < HEAD_DIM; ++d) m += (float)src[d];
        m *= (1.0f / HEAD_DIM);
        float var = 0.f;
        for (int d = 0; d < HEAD_DIM; ++d) { float u = (float)src[d] - m; var += u * u; }
        var *= (1.0f / HEAD_DIM);
        float r = rsqrtf(var + 1e-5f);
        long ob = ((long)((b * HEADS + h) * HEAD_DIM)) * N_TOK + n;
        for (int d = 0; d < HEAD_DIM; ++d)
            dst[ob + (long)d * N_TOK] = (h16)(((float)src[d] - m) * r * g[d] + be[d]);
    }
}

// FNO: pruned DFT basis matrices (computed once on device).
// F [256m][1024p]: m=((kxi*8+ky)*2+ri); kx = kxi<8?kxi:kxi+16. e^{-i theta}.
// Finv [1024p][256m]: (ky==0?1:2)/1024 * Re-projection coefficients of e^{+i theta}.
__global__ void basis_kernel(h16* __restrict__ F, h16* __restrict__ Finv) {
    int idx = blockIdx.x * blockDim.x + threadIdx.x;
    if (idx >= N_MODE_R * N_TOK) return;
    int m = idx >> 10, p = idx & (N_TOK - 1);
    int ri = m & 1, ky = (m >> 1) & 7, kxi = m >> 4;
    int kx = (kxi < 8) ? kxi : kxi + 16;
    int h = p >> 5, w = p & 31;
    int ph = (kx * h + ky * w) & 31;
    float th = 0.19634954084936207f * (float)ph;     // 2*pi/32
    float c = cosf(th), s = sinf(th);
    F[idx] = (h16)(ri ? -s : c);
    float wgt = ((ky == 0) ? 1.0f : 2.0f) * (1.0f / 1024.0f);
    Finv[(long)p * N_MODE_R + m] = (h16)(wgt * (ri ? -s : c));
}

// state f32 [b][p][d] -> xT16 [b][d][p] (for DFT B-operand) and S16 [b*p][d]
__global__ void fno_prep_kernel(const float* __restrict__ S,
                                h16* __restrict__ xT16, h16* __restrict__ S16) {
    int i = blockIdx.x * blockDim.x + threadIdx.x;
    if (i >= M_TOK * EMBED) return;
    float v = S[i];
    S16[i] = (h16)v;
    int d = i & (EMBED - 1), p = (i >> 8) & (N_TOK - 1), b = i >> 18;
    xT16[((long)(b * EMBED + d)) * N_TOK + p] = (h16)v;
}

// per-mode complex channel mixing: YFT[b][o][m] = sum_i W[m](i,o) * XF[b][m][i]
__global__ void modemix_kernel(const float* __restrict__ XF,
                               const float* __restrict__ sw1, const float* __restrict__ sw2,
                               h16* __restrict__ YFT) {
    int b  = blockIdx.x;
    int mm = blockIdx.y;                 // kxi*8 + ky, 0..127
    int kxi = mm >> 3, ky = mm & 7;
    int o = threadIdx.x;
    __shared__ float xr[EMBED], xi[EMBED];
    const float* base = XF + ((long)b * N_MODE_R + mm * 2) * EMBED;
    xr[o] = base[o];
    xi[o] = base[EMBED + o];
    __syncthreads();
    const float* w = (kxi < 8) ? sw1 : sw2;
    int wx = (kxi < 8) ? kxi : kxi - 8;
    float ar = 0.f, ai = 0.f;
    for (int i = 0; i < EMBED; ++i) {
        const float* wp = w + ((((long)i * EMBED + o) * 8 + wx) * 8 + ky) * 2;
        float wr = wp[0], wi = wp[1];
        float cr = xr[i], ci = xi[i];
        ar += cr * wr - ci * wi;
        ai += cr * wi + ci * wr;
    }
    long od = ((long)b * EMBED + o) * N_MODE_R + mm * 2;
    YFT[od]     = (h16)ar;
    YFT[od + 1] = (h16)ai;
}

// out (+)= gatew[b][e] * (S + extra)
__global__ void accum_kernel(float* __restrict__ out, const float* __restrict__ S,
                             const float* __restrict__ extra, const float* __restrict__ gatew,
                             int e, int overwrite, int n) {
    int i = blockIdx.x * blockDim.x + threadIdx.x;
    if (i >= n) return;
    int b = i / (N_TOK * EMBED);
    float w = gatew[b * 4 + e];
    float v = S[i] + (extra ? extra[i] : 0.0f);
    out[i] = overwrite ? (w * v) : (out[i] + w * v);
}

// ---------------------------------------------------------------- host driver
extern "C" void kernel_launch(void* const* d_in, const int* in_sizes, int n_in,
                              void* d_out, int out_size, void* d_ws, size_t ws_size,
                              hipStream_t stream) {
    (void)in_sizes; (void)n_in; (void)out_size; (void)ws_size;
    auto in = [&](int i) { return (const float*)d_in[i]; };

    const float* X   = in(0);
    const float* TXT = in(1);
    const float *G_W1 = in(2), *G_B1 = in(3), *G_W2 = in(4), *G_B2 = in(5);
    // fno: 6+4l {sw1,sw2,cw,cb}; kno: 14+8l {ng,nb,lw,lb,dw,db,pw,pb};
    // mlp: 46+6l {ng,nb,w1,b1,w2,b2}; of: 70+16l {n1g,n1b,qkv_w,qkv_b,lnk_g,lnk_b,
    //                                            lnv_g,lnv_b,proj_w,proj_b,n2g,n2b,mw1,mb1,mw2,mb2}

    char* wsb = (char*)d_ws;
    size_t off = 0;
    auto alloc = [&](size_t bytes) -> void* {
        off = (off + 255) & ~(size_t)255;
        void* p = wsb + off; off += bytes; return p;
    };

    float* gatew   = (float*)alloc(BATCH * 4 * sizeof(float));
    h16*   F16     = (h16*)  alloc((size_t)N_MODE_R * N_TOK * 2);
    h16*   Finv16  = (h16*)  alloc((size_t)N_TOK * N_MODE_R * 2);
    float* S       = (float*)alloc((size_t)M_TOK * EMBED * 4);
    h16* cwT[2];   for (int l = 0; l < 2; ++l) cwT[l]  = (h16*)alloc((size_t)EMBED * EMBED * 2);
    h16* lwT[4], *dwT[4], *pwT[4];
    for (int l = 0; l < 4; ++l) {
        lwT[l] = (h16*)alloc((size_t)KNO_LIFT * EMBED * 2);
        dwT[l] = (h16*)alloc((size_t)KNO_LIFT * KNO_LIFT * 2);
        pwT[l] = (h16*)alloc((size_t)EMBED * KNO_LIFT * 2);
    }
    h16* m1T[4], *m2T[4];
    for (int l = 0; l < 4; ++l) {
        m1T[l] = (h16*)alloc((size_t)MLP_HID * EMBED * 2);
        m2T[l] = (h16*)alloc((size_t)EMBED * MLP_HID * 2);
    }
    h16* qkvT[2], *projT[2], *ow1T[2], *ow2T[2];
    for (int l = 0; l < 2; ++l) {
        qkvT[l]  = (h16*)alloc((size_t)3 * EMBED * EMBED * 2);
        projT[l] = (h16*)alloc((size_t)EMBED * EMBED * 2);
        ow1T[l]  = (h16*)alloc((size_t)MLP_HID * EMBED * 2);
        ow2T[l]  = (h16*)alloc((size_t)EMBED * MLP_HID * 2);
    }
    h16*   ln16   = (h16*)  alloc((size_t)M_TOK * EMBED * 2);
    h16*   bufA   = (h16*)  alloc((size_t)M_TOK * MLP_HID * 2);
    h16*   bufB   = (h16*)  alloc((size_t)M_TOK * KNO_LIFT * 2);
    h16*   xT16   = (h16*)  alloc((size_t)BATCH * EMBED * N_TOK * 2);
    h16*   S16    = (h16*)  alloc((size_t)M_TOK * EMBED * 2);
    float* XF     = (float*)alloc((size_t)BATCH * N_MODE_R * EMBED * 4);
    h16*   YFT    = (h16*)  alloc((size_t)BATCH * EMBED * N_MODE_R * 2);
    float* x1buf  = (float*)alloc((size_t)M_TOK * EMBED * 4);
    h16*   qkv16  = (h16*)  alloc((size_t)M_TOK * 3 * EMBED * 2);
    h16*   kT     = (h16*)  alloc((size_t)BATCH * HEADS * HEAD_DIM * N_TOK * 2);
    h16*   vT     = (h16*)  alloc((size_t)BATCH * HEADS * HEAD_DIM * N_TOK * 2);
    h16*   ctxT   = (h16*)  alloc((size_t)BATCH * HEADS * HEAD_DIM * HEAD_DIM * 2);
    h16*   attn16 = (h16*)  alloc((size_t)M_TOK * EMBED * 2);

    float* OUT = (float*)d_out;
    const int NTOT = M_TOK * EMBED;

    auto wt = [&](const float* W, h16* Wt, int K, int N) {
        int n = K * N;
        wtconv_kernel<<<(n + 255) / 256, 256, 0, stream>>>(W, Wt, K, N);
    };
    auto gemm = [&](const h16* A, const h16* Bt, const float* bias, const float* resid,
                    float* Cf, h16* Ch, int M, int N, int K, int lda, int ldb, int ldc,
                    long sAb, long sAh, long sBb, long sBh, long sCb, long sCh,
                    int batch, int hdiv, float alpha, int act) {
        dim3 g((M + BM - 1) / BM, (N + BN - 1) / BN, batch);
        wmma_gemm_kernel<<<g, 256, 0, stream>>>(A, Bt, bias, resid, Cf, Ch,
            M, N, K, lda, ldb, ldc, sAb, sAh, sBb, sBh, sCb, sCh, hdiv, alpha, act);
    };

    // ---- one-time prep: weight transposes + trig bases + gate
    for (int l = 0; l < 2; ++l) wt(in(6 + 4 * l + 2), cwT[l], EMBED, EMBED);
    for (int l = 0; l < 4; ++l) {
        wt(in(14 + 8 * l + 2), lwT[l], EMBED, KNO_LIFT);
        wt(in(14 + 8 * l + 4), dwT[l], KNO_LIFT, KNO_LIFT);
        wt(in(14 + 8 * l + 6), pwT[l], KNO_LIFT, EMBED);
    }
    for (int l = 0; l < 4; ++l) {
        wt(in(46 + 6 * l + 2), m1T[l], EMBED, MLP_HID);
        wt(in(46 + 6 * l + 4), m2T[l], MLP_HID, EMBED);
    }
    for (int l = 0; l < 2; ++l) {
        wt(in(70 + 16 * l + 2),  qkvT[l], EMBED, 3 * EMBED);
        wt(in(70 + 16 * l + 8),  projT[l], EMBED, EMBED);
        wt(in(70 + 16 * l + 12), ow1T[l], EMBED, MLP_HID);
        wt(in(70 + 16 * l + 14), ow2T[l], MLP_HID, EMBED);
    }
    basis_kernel<<<(N_MODE_R * N_TOK) / 256, 256, 0, stream>>>(F16, Finv16);
    gate_kernel<<<BATCH, EMBED, 0, stream>>>(X, TXT, G_W1, G_B1, G_W2, G_B2, gatew);

    // ================= Expert 0: FNO =================
    copy_f32_kernel<<<(NTOT + 255) / 256, 256, 0, stream>>>(S, X, NTOT);
    for (int l = 0; l < 2; ++l) {
        const float* sw1 = in(6 + 4 * l + 0);
        const float* sw2 = in(6 + 4 * l + 1);
        const float* cb  = in(6 + 4 * l + 3);
        fno_prep_kernel<<<(NTOT + 255) / 256, 256, 0, stream>>>(S, xT16, S16);
        // forward pruned DFT: XF[b] = F (256x1024) * x_b^T  (per batch)
        gemm(F16, xT16, nullptr, nullptr, XF, nullptr,
             N_MODE_R, EMBED, N_TOK, N_TOK, N_TOK, EMBED,
             0, 0, (long)EMBED * N_TOK, 0, (long)N_MODE_R * EMBED, 0,
             BATCH, 1, 1.0f, 0);
        modemix_kernel<<<dim3(BATCH, 128), EMBED, 0, stream>>>(XF, sw1, sw2, YFT);
        // inverse DFT: x1[b] = Finv (1024x256) * YFT[b]^T
        gemm(Finv16, YFT, nullptr, nullptr, x1buf, nullptr,
             N_TOK, EMBED, N_MODE_R, N_MODE_R, N_MODE_R, EMBED,
             0, 0, (long)EMBED * N_MODE_R, 0, (long)N_TOK * EMBED, 0,
             BATCH, 1, 1.0f, 0);
        // 1x1 conv + x1 residual + GELU -> new state (f32)
        gemm(S16, cwT[l], cb, x1buf, S, nullptr,
             M_TOK, EMBED, EMBED, EMBED, EMBED, EMBED,
             0, 0, 0, 0, 0, 0, 1, 1, 1.0f, 1);
    }
    accum_kernel<<<(NTOT + 255) / 256, 256, 0, stream>>>(OUT, S, X, gatew, 0, 1, NTOT);

    // ================= Expert 1: KNO =================
    copy_f32_kernel<<<(NTOT + 255) / 256, 256, 0, stream>>>(S, X, NTOT);
    for (int l = 0; l < 4; ++l) {
        int base = 14 + 8 * l;
        ln_kernel<<<M_TOK, EMBED, 0, stream>>>(S, in(base + 0), in(base + 1), ln16);
        gemm(ln16, lwT[l], in(base + 3), nullptr, nullptr, bufB,
             M_TOK, KNO_LIFT, EMBED, EMBED, EMBED, KNO_LIFT,
             0, 0, 0, 0, 0, 0, 1, 1, 1.0f, 1);
        gemm(bufB, dwT[l], in(base + 5), nullptr, nullptr, bufA,
             M_TOK, KNO_LIFT, KNO_LIFT, KNO_LIFT, KNO_LIFT, KNO_LIFT,
             0, 0, 0, 0, 0, 0, 1, 1, 1.0f, 1);
        gemm(bufA, pwT[l], in(base + 7), S, S, nullptr,
             M_TOK, EMBED, KNO_LIFT, KNO_LIFT, KNO_LIFT, EMBED,
             0, 0, 0, 0, 0, 0, 1, 1, 1.0f, 0);
    }
    accum_kernel<<<(NTOT + 255) / 256, 256, 0, stream>>>(OUT, S, nullptr, gatew, 1, 0, NTOT);

    // ================= Expert 2: MLP =================
    copy_f32_kernel<<<(NTOT + 255) / 256, 256, 0, stream>>>(S, X, NTOT);
    for (int l = 0; l < 4; ++l) {
        int base = 46 + 6 * l;
        ln_kernel<<<M_TOK, EMBED, 0, stream>>>(S, in(base + 0), in(base + 1), ln16);
        gemm(ln16, m1T[l], in(base + 3), nullptr, nullptr, bufA,
             M_TOK, MLP_HID, EMBED, EMBED, EMBED, MLP_HID,
             0, 0, 0, 0, 0, 0, 1, 1, 1.0f, 1);
        gemm(bufA, m2T[l], in(base + 5), S, S, nullptr,
             M_TOK, EMBED, MLP_HID, MLP_HID, MLP_HID, EMBED,
             0, 0, 0, 0, 0, 0, 1, 1, 1.0f, 0);
    }
    accum_kernel<<<(NTOT + 255) / 256, 256, 0, stream>>>(OUT, S, nullptr, gatew, 2, 0, NTOT);

    // ================= Expert 3: OFormer (Galerkin attention) =================
    copy_f32_kernel<<<(NTOT + 255) / 256, 256, 0, stream>>>(S, X, NTOT);
    for (int l = 0; l < 2; ++l) {
        int base = 70 + 16 * l;
        ln_kernel<<<M_TOK, EMBED, 0, stream>>>(S, in(base + 0), in(base + 1), ln16);
        // qkv projection -> f16
        gemm(ln16, qkvT[l], in(base + 3), nullptr, nullptr, qkv16,
             M_TOK, 3 * EMBED, EMBED, EMBED, EMBED, 3 * EMBED,
             0, 0, 0, 0, 0, 0, 1, 1, 1.0f, 0);
        headln_kernel<<<(BATCH * N_TOK * HEADS + 255) / 256, 256, 0, stream>>>(
            qkv16, in(base + 4), in(base + 5), in(base + 6), in(base + 7), kT, vT);
        // ctx^T[e][d] = sum_n v[n,e] k[n,d]   (batched over b,h)
        gemm(vT, kT, nullptr, nullptr, nullptr, ctxT,
             HEAD_DIM, HEAD_DIM, N_TOK, N_TOK, N_TOK, HEAD_DIM,
             (long)HEADS * HEAD_DIM * N_TOK, (long)HEAD_DIM * N_TOK,
             (long)HEADS * HEAD_DIM * N_TOK, (long)HEAD_DIM * N_TOK,
             (long)HEADS * HEAD_DIM * HEAD_DIM, (long)HEAD_DIM * HEAD_DIM,
             BATCH * HEADS, HEADS, 1.0f, 0);
        // attn = (q @ ctx) * 1/sqrt(64) -> merged token-major f16
        gemm(qkv16, ctxT, nullptr, nullptr, nullptr, attn16,
             N_TOK, HEAD_DIM, HEAD_DIM, 3 * EMBED, HEAD_DIM, EMBED,
             (long)N_TOK * 3 * EMBED, (long)HEAD_DIM,
             (long)HEADS * HEAD_DIM * HEAD_DIM, (long)HEAD_DIM * HEAD_DIM,
             (long)N_TOK * EMBED, (long)HEAD_DIM,
             BATCH * HEADS, HEADS, 0.125f, 0);
        // output projection + residual
        gemm(attn16, projT[l], in(base + 9), S, S, nullptr,
             M_TOK, EMBED, EMBED, EMBED, EMBED, EMBED,
             0, 0, 0, 0, 0, 0, 1, 1, 1.0f, 0);
        // MLP block
        ln_kernel<<<M_TOK, EMBED, 0, stream>>>(S, in(base + 10), in(base + 11), ln16);
        gemm(ln16, ow1T[l], in(base + 13), nullptr, nullptr, bufA,
             M_TOK, MLP_HID, EMBED, EMBED, EMBED, MLP_HID,
             0, 0, 0, 0, 0, 0, 1, 1, 1.0f, 1);
        gemm(bufA, ow2T[l], in(base + 15), S, S, nullptr,
             M_TOK, EMBED, MLP_HID, MLP_HID, MLP_HID, EMBED,
             0, 0, 0, 0, 0, 0, 1, 1, 1.0f, 0);
    }
    accum_kernel<<<(NTOT + 255) / 256, 256, 0, stream>>>(OUT, S, nullptr, gatew, 3, 0, NTOT);
}
